// AttentionBlock_33440615366797
// MI455X (gfx1250) — matrix operand order repaired and checked by reading, not compiled, not run on hardware
//
#include <hip/hip_runtime.h>
#include <stdint.h>

// Problem constants (match reference)
#define BB   4
#define LQn  1024
#define LKn  2048
#define En   512
#define HIDn 512
#define NH   8
#define HDn  64

typedef __attribute__((ext_vector_type(16))) __bf16 v16bf;
typedef __attribute__((ext_vector_type(8)))  __bf16 v8bf;
typedef __attribute__((ext_vector_type(8)))  float  v8f;

// ---------------------------------------------------------------------------
// WMMA fragment loaders (gfx1250 wave32 16-bit layouts, CDNA5 ISA)
// ---------------------------------------------------------------------------
static __device__ __forceinline__ v16bf frag_a(const __bf16* p, int ld) {
  const int lane = threadIdx.x & 31;
  const int r = lane & 15, h = lane >> 4;
  const __bf16* row = p + (long)r * ld;
  v8bf lo = *(const v8bf*)(row + h * 8);
  v8bf hi = *(const v8bf*)(row + 16 + h * 8);
  v16bf out;
#pragma unroll
  for (int i = 0; i < 8; ++i) { out[i] = lo[i]; out[i + 8] = hi[i]; }
  return out;
}

static __device__ __forceinline__ v16bf frag_b(const __bf16* p, int ld) {
  const int lane = threadIdx.x & 31;
  const int r = lane & 15, h = lane >> 4;
  const __bf16* row = p + (long)r * ld;
  v8bf lo = *(const v8bf*)(row + h * 8);
  v8bf hi = *(const v8bf*)(row + 16 + h * 8);
  v16bf out;
#pragma unroll
  for (int i = 0; i < 8; ++i) { out[i] = lo[i]; out[i + 8] = hi[i]; }
  return out;
}

static __device__ __forceinline__ v8f wmma_bf16(v16bf a, v16bf b, v8f c) {
  return __builtin_amdgcn_wmma_f32_16x16x32_bf16(
      false, a, false, b, (short)0, c, false, false);
}

// ---------------------------------------------------------------------------
// Generic WMMA GEMM: for z in grid.z: zb = z % zdiv, zh = z / zdiv
//   C[zb,zh] = scale * (A (MxK) * Bt (NxK)^T) + bias
// Both operands K-contiguous. Output fp32 and/or bf16.
// 128 threads = 4 waves; wave -> 32x32 tile; block -> 64x64 tile.
// ---------------------------------------------------------------------------
__global__ void wmma_gemm_kernel(const __bf16* __restrict__ A,
                                 const __bf16* __restrict__ Bt,
                                 const float* __restrict__ bias,
                                 float* __restrict__ Cf,
                                 __bf16* __restrict__ Cbf,
                                 int M, int N, int K,
                                 int lda, int ldb, int ldc,
                                 long aBatch, long bBatch, long cBatch,
                                 long aHead, long bHead, long cHead,
                                 int zdiv, float scale) {
  const int zb = blockIdx.z % zdiv;
  const int zh = blockIdx.z / zdiv;
  A += (long)zb * aBatch + (long)zh * aHead;
  Bt += (long)zb * bBatch + (long)zh * bHead;
  const long cOff = (long)zb * cBatch + (long)zh * cHead;

  const int wid = threadIdx.x >> 5;
  const int wm = wid >> 1, wn = wid & 1;
  const int m0 = blockIdx.y * 64 + wm * 32;
  const int n0 = blockIdx.x * 64 + wn * 32;
  if (m0 >= M || n0 >= N) return;

  v8f acc00 = {}, acc01 = {}, acc10 = {}, acc11 = {};
  const __bf16* arow0 = A + (long)m0 * lda;
  const __bf16* arow1 = A + (long)(m0 + 16) * lda;
  const __bf16* brow0 = Bt + (long)n0 * ldb;
  const __bf16* brow1 = Bt + (long)(n0 + 16) * ldb;

  for (int k0 = 0; k0 < K; k0 += 32) {
    __builtin_prefetch(arow0 + k0 + 128, 0, 1);  // -> global_prefetch_b8
    __builtin_prefetch(brow0 + k0 + 128, 0, 1);
    v16bf a0 = frag_a(arow0 + k0, lda);
    v16bf a1 = frag_a(arow1 + k0, lda);
    v16bf b0 = frag_b(brow0 + k0, ldb);
    v16bf b1 = frag_b(brow1 + k0, ldb);
    acc00 = wmma_bf16(a0, b0, acc00);
    acc01 = wmma_bf16(a0, b1, acc01);
    acc10 = wmma_bf16(a1, b0, acc10);
    acc11 = wmma_bf16(a1, b1, acc11);
  }

  const int lane = threadIdx.x & 31;
  const int cn = lane & 15, hh = lane >> 4;
  v8f accs[4] = {acc00, acc01, acc10, acc11};
#pragma unroll
  for (int im = 0; im < 2; ++im) {
#pragma unroll
    for (int in = 0; in < 2; ++in) {
      v8f acc = accs[im * 2 + in];
      const int mbase = m0 + im * 16 + hh * 8;
      const int nn = n0 + in * 16 + cn;
      const float bv = bias ? bias[nn] : 0.0f;
#pragma unroll
      for (int r = 0; r < 8; ++r) {
        const float v = acc[r] * scale + bv;
        const long idx = cOff + (long)(mbase + r) * ldc + nn;
        if (Cf) Cf[idx] = v;
        if (Cbf) Cbf[idx] = (__bf16)v;
      }
    }
  }
}

// ---------------------------------------------------------------------------
// Reduction helpers (block = 256 = 8 waves)
// ---------------------------------------------------------------------------
static __device__ __forceinline__ float block_sum256(float v, float* sh) {
#pragma unroll
  for (int o = 16; o > 0; o >>= 1) v += __shfl_xor(v, o, 32);
  if ((threadIdx.x & 31) == 0) sh[threadIdx.x >> 5] = v;
  __syncthreads();
  float s = 0.0f;
#pragma unroll
  for (int i = 0; i < 8; ++i) s += sh[i];
  __syncthreads();
  return s;
}

static __device__ __forceinline__ float block_max256(float v, float* sh) {
#pragma unroll
  for (int o = 16; o > 0; o >>= 1) v = fmaxf(v, __shfl_xor(v, o, 32));
  if ((threadIdx.x & 31) == 0) sh[threadIdx.x >> 5] = v;
  __syncthreads();
  float m = sh[0];
#pragma unroll
  for (int i = 1; i < 8; ++i) m = fmaxf(m, sh[i]);
  __syncthreads();
  return m;
}

// ---------------------------------------------------------------------------
// Elementwise kernels
// ---------------------------------------------------------------------------
__global__ void mask_queries_kernel(const float* __restrict__ x,
                                    const int* __restrict__ q_len,
                                    float* __restrict__ of,
                                    __bf16* __restrict__ obf, long n) {
  const long i = (long)blockIdx.x * blockDim.x + threadIdx.x;
  if (i >= n) return;
  const long row = i / En;
  const int b = (int)(row / LQn), l = (int)(row % LQn);
  const float v = (l < q_len[b]) ? x[i] : 0.0f;
  of[i] = v;
  obf[i] = (__bf16)v;
}

__global__ void cvt_bf16_kernel(const float* __restrict__ x,
                                __bf16* __restrict__ o, long n) {
  const long i = (long)blockIdx.x * blockDim.x + threadIdx.x;
  if (i < n) o[i] = (__bf16)x[i];
}

// ---------------------------------------------------------------------------
// 64x64 LDS-tiled transpose: dst[b][c][k] = src[b][k][c].
// Global->LDS fill uses the CDNA5 async-to-LDS path (ASYNCcnt).
// Block = 256 threads; grid = (L/64, C/64, B).
// ---------------------------------------------------------------------------
#define TT 64
__global__ void transpose_bf_async_kernel(const __bf16* __restrict__ src,
                                          __bf16* __restrict__ dst,
                                          int L, int C) {
  __shared__ __bf16 tile[TT][TT + 16];  // row stride 160B (16B aligned, bank spread)
  const int b = blockIdx.z;
  const int k0 = blockIdx.x * TT;  // L index
  const int c0 = blockIdx.y * TT;  // C index
  const int t = threadIdx.x;
  const long base = (long)b * L * C;

#pragma unroll
  for (int it = 0; it < 2; ++it) {
    const int chunk = t + it * 256;          // 0..511
    const int r = chunk >> 3;                // 0..63 row in tile (L)
    const int cc = (chunk & 7) * 8;          // 0..56 col chunk (C)
    const __bf16* gp = src + base + (long)(k0 + r) * C + (c0 + cc);
    const uint32_t laddr = (uint32_t)(uintptr_t)(&tile[r][cc]);
    asm volatile("global_load_async_to_lds_b128 %0, %1, off"
                 :: "v"(laddr), "v"((unsigned long long)(uintptr_t)gp)
                 : "memory");
  }
  asm volatile("s_wait_asynccnt 0x0" ::: "memory");
  __syncthreads();

#pragma unroll
  for (int it = 0; it < 2; ++it) {
    const int chunk = t + it * 256;
    const int cc = chunk >> 3;               // 0..63 output row (C)
    const int rr = (chunk & 7) * 8;          // 0..56 output col chunk (L)
    v8bf val;
#pragma unroll
    for (int i = 0; i < 8; ++i) val[i] = tile[rr + i][cc];
    *(v8bf*)(dst + base + (long)(c0 + cc) * L + (k0 + rr)) = val;
  }
}

// ---------------------------------------------------------------------------
// Multi-head softmax over k for one (b,q) row; loops HC heads (stride hStride),
// writes bf16 probs, and writes/accumulates the head-mean weights once
// (no per-head read-modify-write when HC == NH).
// ---------------------------------------------------------------------------
__global__ void softmax_mh_kernel(const float* __restrict__ S,
                                  __bf16* __restrict__ P,
                                  float* __restrict__ Wacc,
                                  const int* __restrict__ q_len,
                                  int Lq, int Lk, int maskK, float wscale,
                                  int HC, long hStride, int firstChunk) {
  const int q = blockIdx.x, b = blockIdx.y;
  const long roff = ((long)b * Lq + q) * Lk;
  const int ql = q_len[b];
  const int t = threadIdx.x;

  if (q >= ql) {  // invalid query row: zero probs (and weights on first chunk)
    for (int h = 0; h < HC; ++h) {
      __bf16* prow = P + (long)h * hStride + roff;
      for (int k = t; k < Lk; k += 256) prow[k] = (__bf16)0.0f;
    }
    if (Wacc && firstChunk)
      for (int k = t; k < Lk; k += 256) Wacc[roff + k] = 0.0f;
    return;
  }
  const int kl = maskK ? (ql < Lk ? ql : Lk) : Lk;
  __shared__ float sh[8];
  float wsum[8];
#pragma unroll
  for (int i = 0; i < 8; ++i) wsum[i] = 0.0f;

  for (int h = 0; h < HC; ++h) {
    const float* srow = S + (long)h * hStride + roff;
    __bf16* prow = P + (long)h * hStride + roff;
    float m = -3.4e38f;
    for (int k = t; k < kl; k += 256) m = fmaxf(m, srow[k]);
    const float M = block_max256(m, sh);
    float s = 0.0f;
    for (int k = t; k < kl; k += 256) s += __expf(srow[k] - M);
    const float inv = 1.0f / block_sum256(s, sh);
    int seg = 0;
    for (int k = t; k < Lk; k += 256, ++seg) {
      const float p = (k < kl) ? __expf(srow[k] - M) * inv : 0.0f;
      prow[k] = (__bf16)p;
      wsum[seg] += p;
    }
  }
  if (Wacc) {
    int seg = 0;
    for (int k = t; k < Lk; k += 256, ++seg) {
      float v = wsum[seg] * wscale;
      if (!firstChunk) v += Wacc[roff + k];
      Wacc[roff + k] = v;
    }
  }
}

// ---------------------------------------------------------------------------
// y = LN(silu(a + b)) * g + beta ; optional row masking; optional bf16 mirror.
// Row length = 512, block = 256 threads (2 elems/thread).
// ---------------------------------------------------------------------------
__global__ void silu_add_ln_kernel(const float* __restrict__ A,
                                   const float* __restrict__ Bv,
                                   const float* __restrict__ g,
                                   const float* __restrict__ be,
                                   const int* __restrict__ q_len,
                                   int maskOut,
                                   float* __restrict__ Of,
                                   __bf16* __restrict__ Obf) {
  const int row = blockIdx.x;
  const int b = row / LQn, q = row % LQn;
  const long off = (long)row * En;
  const int t = threadIdx.x;
  __shared__ float sh[8];

  float x[2];
#pragma unroll
  for (int i = 0; i < 2; ++i) {
    const int e = t + i * 256;
    const float v = A[off + e] + Bv[off + e];
    x[i] = v / (1.0f + __expf(-v));  // silu
  }
  const float mu = block_sum256(x[0] + x[1], sh) * (1.0f / En);
  const float d0 = x[0] - mu, d1 = x[1] - mu;
  const float var = block_sum256(d0 * d0 + d1 * d1, sh) * (1.0f / En);
  const float r = rsqrtf(var + 1e-5f);
  const bool zero = maskOut && (q >= q_len[b]);
#pragma unroll
  for (int i = 0; i < 2; ++i) {
    const int e = t + i * 256;
    const float y = zero ? 0.0f : (x[i] - mu) * r * g[e] + be[e];
    Of[off + e] = y;
    if (Obf) Obf[off + e] = (__bf16)y;
  }
}

// ---------------------------------------------------------------------------
// Host orchestration
// ---------------------------------------------------------------------------
extern "C" void kernel_launch(void* const* d_in, const int* in_sizes, int n_in,
                              void* d_out, int out_size, void* d_ws, size_t ws_size,
                              hipStream_t stream) {
  const float* queries = (const float*)d_in[0];
  const float* keys    = (const float*)d_in[1];
  const int*   q_len   = (const int*)d_in[2];
  const float* cWq = (const float*)d_in[3];  const float* cbq = (const float*)d_in[4];
  const float* cWk = (const float*)d_in[5];  const float* cbk = (const float*)d_in[6];
  const float* cWv = (const float*)d_in[7];  const float* cbv = (const float*)d_in[8];
  const float* cWo = (const float*)d_in[9];  const float* cbo = (const float*)d_in[10];
  const float* sWq = (const float*)d_in[11]; const float* sbq = (const float*)d_in[12];
  const float* sWk = (const float*)d_in[13]; const float* sbk = (const float*)d_in[14];
  const float* sWv = (const float*)d_in[15]; const float* sbv = (const float*)d_in[16];
  const float* sWo = (const float*)d_in[17]; const float* sbo = (const float*)d_in[18];
  const float* Wf  = (const float*)d_in[19]; const float* bfv = (const float*)d_in[20];
  const float* g_cross = (const float*)d_in[21]; const float* b_cross = (const float*)d_in[22];
  const float* g_ffn   = (const float*)d_in[23]; const float* b_ffn   = (const float*)d_in[24];
  const float* g_self  = (const float*)d_in[25]; const float* b_self  = (const float*)d_in[26];
  (void)in_sizes; (void)n_in; (void)out_size;

  float* out_q = (float*)d_out;                    // [B, LQ, HID]
  float* out_w = out_q + (size_t)BB * LQn * HIDn;  // [B, LQ, LK]

  char* wsp = (char*)d_ws;
  auto alloc = [&](size_t bytes) -> void* {
    void* p = (void*)wsp;
    wsp += (bytes + 255) & ~(size_t)255;
    return p;
  };
  const size_t nQE = (size_t)BB * LQn * En;
  const size_t nKE = (size_t)BB * LKn * En;
  const size_t nWW = (size_t)En * En;

  float*  qmask_f = (float*)alloc(nQE * 4);
  __bf16* q_bf    = (__bf16*)alloc(nQE * 2);
  __bf16* k_bf    = (__bf16*)alloc(nKE * 2);
  __bf16* wb[9];
  for (int i = 0; i < 9; ++i) wb[i] = (__bf16*)alloc(nWW * 2);
  __bf16* cWq_b = wb[0]; __bf16* cWk_b = wb[1]; __bf16* cWv_b = wb[2];
  __bf16* cWo_b = wb[3]; __bf16* sWq_b = wb[4]; __bf16* sWk_b = wb[5];
  __bf16* sWv_b = wb[6]; __bf16* sWo_b = wb[7]; __bf16* Wf_b  = wb[8];
  __bf16* Qp  = (__bf16*)alloc(nQE * 2);
  __bf16* Kp  = (__bf16*)alloc(nKE * 2);
  __bf16* Vp  = (__bf16*)alloc(nKE * 2);
  __bf16* Vt  = (__bf16*)alloc(nKE * 2);
  __bf16* AO  = (__bf16*)alloc(nQE * 2);
  float*  crossO = (float*)alloc(nQE * 4);
  float*  x1f = (float*)alloc(nQE * 4);
  __bf16* x1b = (__bf16*)alloc(nQE * 2);
  float*  ffn = (float*)alloc(nQE * 4);
  float*  x2f = (float*)alloc(nQE * 4);
  __bf16* x2b = (__bf16*)alloc(nQE * 2);
  __bf16* Qp2 = (__bf16*)alloc(nQE * 2);
  __bf16* Kp2 = (__bf16*)alloc(nQE * 2);
  __bf16* Vp2 = (__bf16*)alloc(nQE * 2);
  __bf16* Vt2 = (__bf16*)alloc(nQE * 2);
  __bf16* SO  = (__bf16*)alloc(nQE * 2);
  float*  selfO = (float*)alloc(nQE * 4);

  // Head-chunked score/prob buffers: pick largest HC (heads per pass) that
  // fits in the remaining workspace (deterministic: depends only on ws_size).
  const size_t used = (size_t)(wsp - (char*)d_ws);
  const size_t avail = (ws_size > used) ? ws_size - used : 0;
  const size_t perHead = (size_t)BB * LQn * LKn * 6 + 1024;  // S fp32 + P bf16
  int HC = NH;
  while (HC > 1 && (size_t)HC * perHead > avail) HC >>= 1;
  float*  S = (float*)alloc((size_t)HC * BB * LQn * LKn * 4);
  __bf16* P = (__bf16*)alloc((size_t)HC * BB * LQn * LKn * 2);

  auto gemm = [&](const __bf16* A, const __bf16* Bt, const float* bias,
                  float* Cf, __bf16* Cbf, int M, int N, int K,
                  int lda, int ldb, int ldc,
                  long aB, long bB, long cB,
                  long aH, long bH, long cH,
                  float scale, int nz, int zdiv) {
    dim3 grid((N + 63) / 64, (M + 63) / 64, nz);
    wmma_gemm_kernel<<<grid, 128, 0, stream>>>(A, Bt, bias, Cf, Cbf, M, N, K,
                                               lda, ldb, ldc, aB, bB, cB,
                                               aH, bH, cH, zdiv, scale);
  };

  // ---- 0. convert inputs/weights to bf16 ----------------------------------
  mask_queries_kernel<<<(nQE + 255) / 256, 256, 0, stream>>>(queries, q_len, qmask_f, q_bf, nQE);
  cvt_bf16_kernel<<<(nKE + 255) / 256, 256, 0, stream>>>(keys, k_bf, nKE);
  const float* wsrc[9] = {cWq, cWk, cWv, cWo, sWq, sWk, sWv, sWo, Wf};
  for (int i = 0; i < 9; ++i)
    cvt_bf16_kernel<<<(nWW + 255) / 256, 256, 0, stream>>>(wsrc[i], wb[i], nWW);

  // ---- 1. cross-attention Q/K/V projections (bf16 out) --------------------
  gemm(q_bf, cWq_b, cbq, nullptr, Qp, BB * LQn, En, En, En, En, En,
       0, 0, 0, 0, 0, 0, 1.0f, 1, 1);
  gemm(k_bf, cWk_b, cbk, nullptr, Kp, BB * LKn, En, En, En, En, En,
       0, 0, 0, 0, 0, 0, 1.0f, 1, 1);
  gemm(k_bf, cWv_b, cbv, nullptr, Vp, BB * LKn, En, En, En, En, En,
       0, 0, 0, 0, 0, 0, 1.0f, 1, 1);
  transpose_bf_async_kernel<<<dim3(LKn / TT, En / TT, BB), 256, 0, stream>>>(Vp, Vt, LKn, En);

  // ---- 2. cross-attention, HC heads per pass ------------------------------
  const float iscale = 0.125f;  // 1/sqrt(64)
  const long hS_cross = (long)BB * LQn * LKn;
  for (int hb = 0; hb < NH; hb += HC) {
    gemm(Qp + hb * HDn, Kp + hb * HDn, nullptr, S, nullptr,
         LQn, LKn, HDn, En, En, LKn,
         (long)LQn * En, (long)LKn * En, (long)LQn * LKn,
         HDn, HDn, hS_cross, iscale, BB * HC, BB);
    softmax_mh_kernel<<<dim3(LQn, BB), 256, 0, stream>>>(
        S, P, out_w, q_len, LQn, LKn, 0, 1.0f / NH, HC, hS_cross, hb == 0);
    gemm(P, Vt + (long)hb * HDn * LKn, nullptr, nullptr, AO + hb * HDn,
         LQn, HDn, LKn, LKn, LKn, En,
         (long)LQn * LKn, (long)En * LKn, (long)LQn * En,
         hS_cross, (long)HDn * LKn, HDn, 1.0f, BB * HC, BB);
  }
  gemm(AO, cWo_b, cbo, crossO, nullptr, BB * LQn, En, En, En, En, En,
       0, 0, 0, 0, 0, 0, 1.0f, 1, 1);

  // ---- 3. residual + silu + LN, FFN, residual + silu + LN -----------------
  silu_add_ln_kernel<<<BB * LQn, 256, 0, stream>>>(crossO, qmask_f, g_cross, b_cross,
                                                   q_len, 0, x1f, x1b);
  gemm(x1b, Wf_b, bfv, ffn, nullptr, BB * LQn, HIDn, En, En, En, HIDn,
       0, 0, 0, 0, 0, 0, 1.0f, 1, 1);
  silu_add_ln_kernel<<<BB * LQn, 256, 0, stream>>>(x1f, ffn, g_ffn, b_ffn,
                                                   q_len, 1, x2f, x2b);

  // ---- 4. self-attention --------------------------------------------------
  gemm(x2b, sWq_b, sbq, nullptr, Qp2, BB * LQn, HIDn, HIDn, HIDn, HIDn, HIDn,
       0, 0, 0, 0, 0, 0, 1.0f, 1, 1);
  gemm(x2b, sWk_b, sbk, nullptr, Kp2, BB * LQn, HIDn, HIDn, HIDn, HIDn, HIDn,
       0, 0, 0, 0, 0, 0, 1.0f, 1, 1);
  gemm(x2b, sWv_b, sbv, nullptr, Vp2, BB * LQn, HIDn, HIDn, HIDn, HIDn, HIDn,
       0, 0, 0, 0, 0, 0, 1.0f, 1, 1);
  transpose_bf_async_kernel<<<dim3(LQn / TT, HIDn / TT, BB), 256, 0, stream>>>(Vp2, Vt2, LQn, HIDn);

  const long hS_self = (long)BB * LQn * LQn;
  for (int hb = 0; hb < NH; hb += HC) {
    gemm(Qp2 + hb * HDn, Kp2 + hb * HDn, nullptr, S, nullptr,
         LQn, LQn, HDn, HIDn, HIDn, LQn,
         (long)LQn * HIDn, (long)LQn * HIDn, (long)LQn * LQn,
         HDn, HDn, hS_self, iscale, BB * HC, BB);
    softmax_mh_kernel<<<dim3(LQn, BB), 256, 0, stream>>>(
        S, P, nullptr, q_len, LQn, LQn, 1, 0.0f, HC, hS_self, 0);
    gemm(P, Vt2 + (long)hb * HDn * LQn, nullptr, nullptr, SO + hb * HDn,
         LQn, HDn, LQn, LQn, LQn, HIDn,
         (long)LQn * LQn, (long)HIDn * LQn, (long)LQn * HIDn,
         hS_self, (long)HDn * LQn, HDn, 1.0f, BB * HC, BB);
  }
  gemm(SO, sWo_b, sbo, selfO, nullptr, BB * LQn, HIDn, HIDn, HIDn, HIDn, HIDn,
       0, 0, 0, 0, 0, 0, 1.0f, 1, 1);

  // ---- 5. final residual + silu + LN -> out_q (masked) --------------------
  silu_add_ln_kernel<<<BB * LQn, 256, 0, stream>>>(x2f, selfO, g_self, b_self,
                                                   q_len, 1, out_q, nullptr);
}
